// PatchLevelKL_14989435863423
// MI455X (gfx1250) — matrix operand order repaired and checked by reading, not compiled, not run on hardware
//
#include <hip/hip_runtime.h>
#include <hip/hip_bf16.h>
#include <cstdint>

// ---------------------------------------------------------------------------
// Problem constants (from reference): B=1024, T=P=197, D=64, Z_DIM=2
// ---------------------------------------------------------------------------
#define NB   1024
#define NTOK 197
#define NQ   196           // 197-1
#define DIM  64

typedef __attribute__((ext_vector_type(16))) _Float16 v16h;
typedef __attribute__((ext_vector_type(8)))  float    v8f;

__device__ __forceinline__ v8f wmma_f32_16x16x32_f16(v16h a, v16h b, v8f c) {
  // 8 args: (neg_a, A, neg_b, B, c_mod, C, reuse_a, reuse_b)
  return __builtin_amdgcn_wmma_f32_16x16x32_f16(false, a, false, b, (short)0, c,
                                                false, false);
}

// ------------------------- deterministic RNG (hash) -------------------------
__device__ __forceinline__ uint32_t pcg_hash(uint32_t v) {
  v = v * 747796405u + 2891336453u;
  v = ((v >> ((v >> 28u) + 4u)) ^ v) * 277803737u;
  return (v >> 22u) ^ v;
}
__device__ __forceinline__ float hash_normal(uint32_t a, uint32_t seed) {
  uint32_t h1 = pcg_hash(a * 0x9E3779B9u ^ seed);
  uint32_t h2 = pcg_hash(h1 ^ 0x85EBCA6Bu);
  float u1 = (float)((h1 >> 8) + 1u) * (1.0f / 16777217.0f); // (0,1]
  float u2 = (float)(h2 >> 8) * (1.0f / 16777216.0f);        // [0,1)
  return sqrtf(-2.0f * __logf(u1)) * __cosf(6.28318530717958647f * u2);
}

__device__ __forceinline__ float softplus_f(float x) {
  return (x > 20.0f) ? x : log1pf(__expf(x));
}
__device__ __forceinline__ float sigmoid_f(float x) {
  return 1.0f / (1.0f + __expf(-x));
}

// ===========================================================================
// Kernel 1: fused dual argmax over sim[:,1:,1:]  (one pass over 157 MB)
//   idxP[b,t] = argmax_p sim[b,t+1,p+1]   (row argmax, axis=2)
//   idxI[b,p] = argmax_t sim[b,t+1,p+1]   (col argmax, axis=1)
// Block = 256 threads (8 waves) per batch b. Wave w owns rows t = w, w+8,...
// Lane strides columns (p = lane + 32k) -> fully coalesced 128B reads.
// Row argmax: barrier-free shfl_xor reduction. Col argmax: per-thread
// registers, merged across waves through LDS at the end.
// ===========================================================================
__global__ __launch_bounds__(256)
void dual_argmax_kernel(const float* __restrict__ sim,
                        int* __restrict__ idxP, int* __restrict__ idxI)
{
  __shared__ float pv[8 * NQ];
  __shared__ int   pi[8 * NQ];

  const int b    = blockIdx.x;
  const int tid  = threadIdx.x;
  const int wave = tid >> 5;
  const int lane = tid & 31;
  const float* base = sim + (size_t)b * NTOK * NTOK;

  float cmax[7];
  int   cidx[7];
#pragma unroll
  for (int k = 0; k < 7; ++k) { cmax[k] = -INFINITY; cidx[k] = 0; }

  for (int t = wave; t < NQ; t += 8) {
    const float* row = base + (size_t)(t + 1) * NTOK + 1;
    float bv = -INFINITY; int bp = 0;
#pragma unroll
    for (int k = 0; k < 7; ++k) {
      const int p = lane + 32 * k;
      if (p < NQ) {
        const float v = row[p];
        if (v > bv)      { bv = v; bp = p; }         // first-max tie rule
        if (v > cmax[k]) { cmax[k] = v; cidx[k] = t; }
      }
    }
    // in-wave argmax reduction (wave32)
#pragma unroll
    for (int off = 16; off > 0; off >>= 1) {
      const float ov = __shfl_xor(bv, off, 32);
      const int   oi = __shfl_xor(bp, off, 32);
      if (ov > bv || (ov == bv && oi < bp)) { bv = ov; bp = oi; }
    }
    if (lane == 0) idxP[(size_t)b * NQ + t] = bp;
  }

  // publish per-wave column partials
#pragma unroll
  for (int k = 0; k < 7; ++k) {
    const int p = lane + 32 * k;
    if (p < NQ) { pv[wave * NQ + p] = cmax[k]; pi[wave * NQ + p] = cidx[k]; }
  }
  __syncthreads();

  if (tid < NQ) {
    float bv = pv[tid]; int bt = pi[tid];
#pragma unroll
    for (int w = 1; w < 8; ++w) {
      const float v = pv[w * NQ + tid];
      const int   i2 = pi[w * NQ + tid];
      if (v > bv || (v == bv && i2 < bt)) { bv = v; bt = i2; }
    }
    idxI[(size_t)b * NQ + tid] = bt;
  }
}

// ===========================================================================
// Kernel 2: WMMA MLP + Gaussian log-prob KL.
// One wave = one 16-row tile (B*NQ = 200704 rows = 12544 tiles exactly).
// Two pipelines per tile: pipe0 = direct rows (x @ W*), pipe1 = gathered rows.
// Layer1 (64->64): 4 N-tiles x 2 K-steps of v_wmma_f32_16x16x32_f16.
// Layer2 (64->4, padded to 16): 2 K-steps. h staged via per-wave LDS.
// Weights cached in LDS transposed ([n][k]) so B-fragments are contiguous.
// ===========================================================================
__global__ __launch_bounds__(256)
void mlp_kl_kernel(const float* __restrict__ xAlign,   // (B,197,64): rows t+1
                   const float* __restrict__ gAlign,   // (B,197,64): gathered
                   const int*   __restrict__ gIdx,     // (B,196)
                   const float* __restrict__ W1x, const float* __restrict__ b1x,
                   const float* __restrict__ W2x, const float* __restrict__ b2x,
                   const float* __restrict__ W1g, const float* __restrict__ b1g,
                   const float* __restrict__ W2g, const float* __restrict__ b2g,
                   float* __restrict__ out, uint32_t seed)
{
  __shared__ _Float16 sW1x[DIM * DIM];   // transposed: [n][k]
  __shared__ _Float16 sW1g[DIM * DIM];
  __shared__ _Float16 sW2x[16 * DIM];    // padded N=16, transposed: [n][k]
  __shared__ _Float16 sW2g[16 * DIM];
  __shared__ _Float16 sH [8][16 * DIM];  // per-wave h tile (16x64)
  __shared__ float    sPx[8][16 * 16];   // per-wave layer2 out, pipe0
  __shared__ float    sPg[8][16 * 16];   // per-wave layer2 out, pipe1

  const int tid = threadIdx.x;

  // cooperative weight load (transpose into LDS as f16)
  for (int i = tid; i < DIM * DIM; i += 256) {
    const int k = i >> 6, n = i & 63;
    sW1x[n * DIM + k] = (_Float16)W1x[i];
    sW1g[n * DIM + k] = (_Float16)W1g[i];
  }
  for (int i = tid; i < 16 * DIM; i += 256) {
    const int n = i >> 6, k = i & 63;
    sW2x[i] = (_Float16)((n < 4) ? W2x[k * 4 + n] : 0.0f);
    sW2g[i] = (_Float16)((n < 4) ? W2g[k * 4 + n] : 0.0f);
  }
  __syncthreads();

  const int wave = tid >> 5;
  const int lane = tid & 31;
  const int hi   = lane >> 4;   // 0/1 lane-half
  const int lo   = lane & 15;
  const int tile = blockIdx.x * 8 + wave;

  for (int pipe = 0; pipe < 2; ++pipe) {
    // ---- build A fragments (16x64 of inputs, f16) straight from global ----
    const int rA = tile * 16 + lo;       // row owned by this lane (A layout)
    const int bA = rA / NQ;
    const int tA = rA % NQ;
    const float* rowPtr;
    if (pipe == 0) {
      rowPtr = xAlign + ((size_t)bA * NTOK + (tA + 1)) * DIM;
    } else {
      const int gi = gIdx[(size_t)bA * NQ + tA];
      rowPtr = gAlign + ((size_t)bA * NTOK + gi) * DIM;
    }
    v16h a0, a1;
    {
      const int c0 = hi * 8;        // k-step 0: halves 0..7 -> K=c0..c0+7,
#pragma unroll                      //           halves 8..15 -> K=c0+16..c0+23
      for (int j = 0; j < 8; ++j) {
        a0[j]     = (_Float16)rowPtr[c0 + j];
        a0[8 + j] = (_Float16)rowPtr[c0 + 16 + j];
      }
      const int c1 = 32 + hi * 8;   // k-step 1
#pragma unroll
      for (int j = 0; j < 8; ++j) {
        a1[j]     = (_Float16)rowPtr[c1 + j];
        a1[8 + j] = (_Float16)rowPtr[c1 + 16 + j];
      }
    }

    const _Float16* sW1 = (pipe == 0) ? sW1x : sW1g;
    const _Float16* sW2 = (pipe == 0) ? sW2x : sW2g;
    const float* bias1  = (pipe == 0) ? b1x : b1g;
    const float* bias2  = (pipe == 0) ? b2x : b2g;

    // ---------------- layer 1: h = relu(x @ W1 + b1) -----------------------
#pragma unroll
    for (int nt = 0; nt < 4; ++nt) {
      v16h bf0, bf1;
      {
        // B frag: lane owns column n = nt*16+lo; halves 0..7 -> K=hi*8.., 8..15 -> K=16+hi*8..
        const _Float16* wcol = sW1 + (nt * 16 + lo) * DIM + hi * 8;
#pragma unroll
        for (int j = 0; j < 16; ++j) { const int kk = (j < 8) ? j : 8 + j; bf0[j] = wcol[kk]; bf1[j] = wcol[32 + kk]; }
      }
      v8f acc = {};
      acc = wmma_f32_16x16x32_f16(a0, bf0, acc);
      acc = wmma_f32_16x16x32_f16(a1, bf1, acc);

      const float bb = bias1[nt * 16 + lo];
#pragma unroll
      for (int rr = 0; rr < 8; ++rr) {
        float h = acc[rr] + bb;
        h = (h > 0.0f) ? h : 0.0f;
        const int m = rr + hi * 8;                    // C/D row in tile
        sH[wave][m * DIM + nt * 16 + lo] = (_Float16)h;
      }
    }
    __syncthreads();

    // ---------------- layer 2: p = h @ W2pad + b2 --------------------------
    v16h ha0, ha1;
    {
      const _Float16* hrow = &sH[wave][lo * DIM];
      const int c0 = hi * 8;
#pragma unroll
      for (int j = 0; j < 8; ++j) {
        ha0[j] = hrow[c0 + j];       ha0[8 + j] = hrow[c0 + 16 + j];
      }
      const int c1 = 32 + hi * 8;
#pragma unroll
      for (int j = 0; j < 8; ++j) {
        ha1[j] = hrow[c1 + j];       ha1[8 + j] = hrow[c1 + 16 + j];
      }
    }
    v8f pacc = {};
    {
      v16h bf0, bf1;
      const _Float16* wcol = sW2 + lo * DIM + hi * 8;
#pragma unroll
      for (int j = 0; j < 16; ++j) { const int kk = (j < 8) ? j : 8 + j; bf0[j] = wcol[kk]; bf1[j] = wcol[32 + kk]; }
      pacc = wmma_f32_16x16x32_f16(ha0, bf0, pacc);
      pacc = wmma_f32_16x16x32_f16(ha1, bf1, pacc);
    }
    const float bb2 = (lo < 4) ? bias2[lo] : 0.0f;
    float* pdst = (pipe == 0) ? sPx[wave] : sPg[wave];
#pragma unroll
    for (int rr = 0; rr < 8; ++rr) {
      const int m = rr + hi * 8;
      pdst[m * 16 + lo] = pacc[rr] + bb2;
    }
    __syncthreads();
  }

  // ---------------- epilogue: Gaussian log-prob KL (lanes 0..15) -----------
  if (lane < 16) {
    const int r = tile * 16 + lane;
    const float* px = &sPx[wave][lane * 16];
    const float* pg = &sPg[wave][lane * 16];

    const float mux0 = px[0], mux1 = px[1];
    const float sx0  = softplus_f(px[2]) + 1e-7f;
    const float sx1  = softplus_f(px[3]) + 1e-7f;
    const float mug0 = pg[0], mug1 = pg[1];
    const float sg0  = softplus_f(pg[2]) + 1e-7f;
    const float sg1  = softplus_f(pg[3]) + 1e-7f;

    const float e0 = hash_normal((uint32_t)r * 2u + 0u, seed);
    const float e1 = hash_normal((uint32_t)r * 2u + 1u, seed);
    const float z0 = mux0 + sx0 * e0;
    const float z1 = mux1 + sx1 * e1;

    const float HALF_L2PI = 0.918938533204672741f;   // 0.5*log(2*pi)
    // log N(z | mu_x, sg_x): (z-mu_x)/sg_x == eps
    const float lpx = -0.5f * (e0 * e0 + e1 * e1)
                      - __logf(sx0) - __logf(sx1) - 2.0f * HALF_L2PI;
    const float d0 = (z0 - mug0) / sg0;
    const float d1 = (z1 - mug1) / sg1;
    const float lpg = -0.5f * (d0 * d0 + d1 * d1)
                      - __logf(sg0) - __logf(sg1) - 2.0f * HALF_L2PI;

    out[r] = sigmoid_f(sigmoid_f(lpx - lpg));        // double sigmoid per ref
  }
}

// ===========================================================================
// Launch
// ===========================================================================
extern "C" void kernel_launch(void* const* d_in, const int* in_sizes, int n_in,
                              void* d_out, int out_size, void* d_ws, size_t ws_size,
                              hipStream_t stream) {
  const float* post  = (const float*)d_in[0];   // (B,197,64)
  const float* image = (const float*)d_in[1];   // (B,197,64)
  const float* sim   = (const float*)d_in[2];   // (B,197,197)
  const float* tW1   = (const float*)d_in[3];
  const float* tb1   = (const float*)d_in[4];
  const float* tW2   = (const float*)d_in[5];
  const float* tb2   = (const float*)d_in[6];
  const float* iW1   = (const float*)d_in[7];
  const float* ib1   = (const float*)d_in[8];
  const float* iW2   = (const float*)d_in[9];
  const float* ib2   = (const float*)d_in[10];
  float* out = (float*)d_out;

  int* idxP = (int*)d_ws;                       // (B,196)
  int* idxI = idxP + (size_t)NB * NQ;           // (B,196)

  // Pass 1: one read of sim, both argmax directions.
  dual_argmax_kernel<<<NB, 256, 0, stream>>>(sim, idxP, idxI);

  // Pass 2: 12544 tiles / 8 waves per block = 1568 blocks per side.
  const int nTiles  = (NB * NQ) / 16;           // 12544 exactly
  const int nBlocks = nTiles / 8;               // 1568

  // side 0 (kl_post): x = post[:,1:], gathered = image[idxP], weights tW / iW
  mlp_kl_kernel<<<nBlocks, 256, 0, stream>>>(
      post, image, idxP,
      tW1, tb1, tW2, tb2,
      iW1, ib1, iW2, ib2,
      out, 0x1234ABCDu);

  // side 1 (kl_image): x = image[:,1:], gathered = post[idxI], weights iW / tW
  mlp_kl_kernel<<<nBlocks, 256, 0, stream>>>(
      image, post, idxI,
      iW1, ib1, iW2, ib2,
      tW1, tb1, tW2, tb2,
      out + (size_t)NB * NQ, 0x5678EF01u);
}
